// Decoder_75548474736723
// MI455X (gfx1250) — matrix operand order repaired and checked by reading, not compiled, not run on hardware
//
#include <hip/hip_runtime.h>
#include <hip/hip_bf16.h>

// ---------------------------------------------------------------- constants
#define LAYERS 4
#define SEQ    2048
#define DM     1024
#define DFF    4096
#define NH     16
#define DH     64
#define PLEN   16
#define CS     16                 // mps chunk size
#define NC     (SEQ / CS)         // 128 chunks
#define STATE  (NH * DH * PLEN)   // 16384 per-chunk state entries (both mps)

typedef __attribute__((ext_vector_type(16))) __bf16 v16bf;
typedef __attribute__((ext_vector_type(8)))  float  v8f;

__device__ __forceinline__ unsigned short f2bfu(float f) {
  unsigned u = __builtin_bit_cast(unsigned, f);
  unsigned r = u + 0x7FFFu + ((u >> 16) & 1u);      // round-to-nearest-even
  return (unsigned short)(r >> 16);
}

// ---------------------------------------------------------------- embedding
__global__ __launch_bounds__(256) void k_embed(const int* __restrict__ x,
                                               const float* __restrict__ emb,
                                               float* __restrict__ h) {
  int i = blockIdx.x * 256 + threadIdx.x;   // SEQ*DM total
  int t = i >> 10, d = i & 1023;
  h[i] = 32.0f * emb[(size_t)x[t] * DM + d];   // sqrt(1024)=32
}

__global__ __launch_bounds__(256) void k_addpos(const float* __restrict__ h,
                                                const float* __restrict__ pos,
                                                float* __restrict__ xe) {
  int i = blockIdx.x * 256 + threadIdx.x;
  xe[i] = h[i] + 32.0f * pos[i];
}

// ---------------------------------------------------------------- WMMA GEMM
// C[M,N] = act( (A[M,K] x B[K,N] + bias[N]) * alpha ), fp32 in, bf16 WMMA math.
// LDS tiles are stored pre-swizzled in WMMA fragment order so every lane's
// 16-element fragment is 32 contiguous bytes (2x ds_load_b128).
#define TM 128
#define TN 128
#define TK 64

// fragment-order LDS index helpers (bf16 elements)
//  A tile: 128 rows x 64 k.  sub-step ks=kk>>5, kl=kk&31.
//   row-tile i=r>>4, lane=(r&15)+16*(kl>>4), elem e=kl&15
//  B tile: 64 k x 128 cols.  ks=r>>5, rl=r&31.
//   col-tile j=c>>4, lane=(c&15)+16*(rl>>4), elem e=rl&15
__device__ __forceinline__ int a_swz(int r, int kk) {
  int ks = kk >> 5, kl = kk & 31;
  int lane = (r & 15) + 16 * (kl >> 4);
  int e = kl & 15;
  return (((ks * 8) + (r >> 4)) * 32 + lane) * 16 + e;
}
__device__ __forceinline__ int b_swz(int r, int c) {
  int ks = r >> 5, rl = r & 31;
  int lane = (c & 15) + 16 * (rl >> 4);
  int e = rl & 15;
  return (((ks * 8) + (c >> 4)) * 32 + lane) * 16 + e;
}

__global__ __launch_bounds__(256) void k_gemm(const float* __restrict__ A,
                                              const float* __restrict__ B,
                                              const float* __restrict__ bias,
                                              float* __restrict__ C,
                                              int M, int N, int K,
                                              float alpha, int relu) {
  __shared__ __align__(32) unsigned short sA[2 * 8 * 32 * 16];  // 16 KB
  __shared__ __align__(32) unsigned short sB[2 * 8 * 32 * 16];  // 16 KB

  const int tid  = threadIdx.x;
  const int lane = tid & 31;
  const int wave = tid >> 5;     // 0..7
  const int wr   = wave >> 2;    // 0..1 : 64-row group
  const int wc   = wave & 3;     // 0..3 : 32-col group
  const int half = lane >> 4;
  const int l16  = lane & 15;
  const int bm   = blockIdx.x * TM;
  const int bn   = blockIdx.y * TN;

  v8f acc[4][2];
  #pragma unroll
  for (int i = 0; i < 4; ++i)
    #pragma unroll
    for (int j = 0; j < 2; ++j)
      #pragma unroll
      for (int e = 0; e < 8; ++e) acc[i][j][e] = 0.0f;

  for (int k0 = 0; k0 < K; k0 += TK) {
    // ---- stage A: 128x64 fp32 -> bf16, paired along k (one b32 LDS store)
    #pragma unroll
    for (int p = tid; p < (TM * TK) / 2; p += 256) {      // 4096 pairs
      int r = p >> 5;                  // 32 k-pairs per row
      int c = (p & 31) * 2;
      const float2 f2 = *reinterpret_cast<const float2*>(
          A + (size_t)(bm + r) * K + k0 + c);
      unsigned u = (unsigned)f2bfu(f2.x) | ((unsigned)f2bfu(f2.y) << 16);
      *reinterpret_cast<unsigned*>(&sA[a_swz(r, c)]) = u;  // e even, e+1 next
    }
    // ---- stage B: 64x128 fp32 -> bf16, paired along k-rows (one b32 store)
    #pragma unroll
    for (int p = tid; p < (TK * TN) / 2; p += 256) {      // 4096 pairs
      int c = p & 127;
      int r = (p >> 7) * 2;
      float f0 = B[(size_t)(k0 + r) * N + bn + c];
      float f1 = B[(size_t)(k0 + r + 1) * N + bn + c];
      unsigned u = (unsigned)f2bfu(f0) | ((unsigned)f2bfu(f1) << 16);
      *reinterpret_cast<unsigned*>(&sB[b_swz(r, c)]) = u;
    }
    if (k0 + TK < K) {   // hint next K tile into cache (global_prefetch_b8)
      __builtin_prefetch(A + (size_t)(bm + (tid >> 1)) * K + k0 + TK, 0, 1);
      __builtin_prefetch(B + (size_t)(k0 + TK + (tid & 63)) * N + bn + ((tid >> 6) << 5), 0, 1);
    }
    __syncthreads();

    #pragma unroll
    for (int ks = 0; ks < 2; ++ks) {
      v16bf bf[2];
      #pragma unroll
      for (int j = 0; j < 2; ++j) {
        int boff = ((ks * 8 + wc * 2 + j) * 32 + lane) * 16;
        bf[j] = *reinterpret_cast<const v16bf*>(&sB[boff]);
      }
      #pragma unroll
      for (int i = 0; i < 4; ++i) {
        int aoff = ((ks * 8 + wr * 4 + i) * 32 + lane) * 16;
        v16bf af = *reinterpret_cast<const v16bf*>(&sA[aoff]);
        acc[i][0] = __builtin_amdgcn_wmma_f32_16x16x32_bf16(
            false, af, false, bf[0], (short)0, acc[i][0], false, false);
        acc[i][1] = __builtin_amdgcn_wmma_f32_16x16x32_bf16(
            false, af, false, bf[1], (short)0, acc[i][1], false, false);
      }
    }
    __syncthreads();
  }

  // C/D layout: VGPR e -> row e + 8*half, col = lane%16
  #pragma unroll
  for (int i = 0; i < 4; ++i)
    #pragma unroll
    for (int j = 0; j < 2; ++j) {
      int col = bn + wc * 32 + j * 16 + l16;
      float bb = bias[col];
      #pragma unroll
      for (int e = 0; e < 8; ++e) {
        int row = bm + wr * 64 + i * 16 + e + 8 * half;
        float vv = (acc[i][j][e] + bb) * alpha;
        if (relu) vv = fmaxf(vv, 0.0f);
        C[(size_t)row * N + col] = vv;
      }
    }
}

// ---------------------------------------------------------------- pack = elu(p.q)+1
__global__ __launch_bounds__(256) void k_pack(const float* __restrict__ pl,
                                              const float* __restrict__ q,
                                              float* __restrict__ pack) {
  int t = blockIdx.x;
  int hh = threadIdx.x >> 4, pp = threadIdx.x & 15;
  const float* qr = q  + (size_t)t  * DM + hh * DH;
  const float* pr = pl + (size_t)pp * DM + hh * DH;
  float a = 0.0f;
  #pragma unroll 8
  for (int d = 0; d < DH; ++d) a += pr[d] * qr[d];
  float e = a > 0.0f ? a : __expf(a) - 1.0f;
  pack[(size_t)t * (NH * PLEN) + hh * PLEN + pp] = e + 1.0f;
}

// ---------------------------------------------------------------- mps chunked scan
// O[c][h][i][j] = sum_{s in chunk c} Y[s,h,i] * Z[s,h,j]
__global__ __launch_bounds__(256) void k_chunk_outer(const float* __restrict__ Y,
                                                     const float* __restrict__ Z,
                                                     float* __restrict__ O,
                                                     int dimY, int dimZ) {
  int idx = blockIdx.x * 256 + threadIdx.x;    // NC*NH*dimY*dimZ
  int j = idx % dimZ;
  int r = idx / dimZ;
  int i = r % dimY; r /= dimY;
  int hh = r & 15;
  int c  = r >> 4;
  int ldY = NH * dimY, ldZ = NH * dimZ;
  int s0 = c * CS;
  float a = 0.0f;
  #pragma unroll
  for (int s = 0; s < CS; ++s)
    a += Y[(size_t)(s0 + s) * ldY + hh * dimY + i] *
         Z[(size_t)(s0 + s) * ldZ + hh * dimZ + j];
  O[idx] = a;
}

// in-place exclusive prefix over chunks: one thread per state entry
__global__ __launch_bounds__(256) void k_chunk_scan(float* __restrict__ O, int nc) {
  int e = blockIdx.x * 256 + threadIdx.x;      // STATE entries
  float run = 0.0f;
  for (int c = 0; c < nc; ++c) {
    size_t off = (size_t)c * STATE + e;
    float t = O[off];
    O[off] = run;
    run += t;
  }
}

// out[t,h,p] = (q[t]·P[chunk] + sum_{s in chunk, s<=t} (q[t]·k[s]) pack[s,p]) / (t+1)
__global__ __launch_bounds__(256) void k_mps1_out(const float* __restrict__ q,
                                                  const float* __restrict__ k,
                                                  const float* __restrict__ pack,
                                                  const float* __restrict__ P,
                                                  float* __restrict__ out) {
  __shared__ float dots[NH][CS];
  int t = blockIdx.x;
  int c = t >> 4, s0 = c << 4, nIn = t - s0 + 1;
  int hh = threadIdx.x >> 4, lo = threadIdx.x & 15;
  {
    int s = s0 + lo;
    float d = 0.0f;
    if (s <= t) {
      const float* qr = q + (size_t)t * DM + hh * DH;
      const float* kr = k + (size_t)s * DM + hh * DH;
      #pragma unroll 8
      for (int dd = 0; dd < DH; ++dd) d += qr[dd] * kr[dd];
    }
    dots[hh][lo] = d;
  }
  __syncthreads();
  const float* qr = q + (size_t)t * DM + hh * DH;
  const float* Pp = P + (size_t)c * STATE + hh * (DH * PLEN) + lo;
  float a = 0.0f;
  #pragma unroll 8
  for (int dd = 0; dd < DH; ++dd) a += qr[dd] * Pp[dd * PLEN];
  for (int si = 0; si < nIn; ++si)
    a += dots[hh][si] * pack[(size_t)(s0 + si) * (NH * PLEN) + hh * PLEN + lo];
  out[(size_t)t * (NH * PLEN) + hh * PLEN + lo] = a / (float)(t + 1);
}

__global__ __launch_bounds__(256) void k_softmax16(float* __restrict__ u) {
  int i = blockIdx.x * 256 + threadIdx.x;      // SEQ*NH rows of 16
  float* r = u + (size_t)i * PLEN;
  float m = -1e30f;
  #pragma unroll
  for (int p = 0; p < PLEN; ++p) m = fmaxf(m, r[p]);
  float e[PLEN], s = 0.0f;
  #pragma unroll
  for (int p = 0; p < PLEN; ++p) { e[p] = __expf(r[p] - m); s += e[p]; }
  float inv = 1.0f / s;
  #pragma unroll
  for (int p = 0; p < PLEN; ++p) r[p] = e[p] * inv;
}

// attn[t,h,d] = (u[t]·P2[chunk][:,d] + sum_{s<=t in chunk} (u[t]·pack[s]) v[s,d]) / (t+1)
__global__ __launch_bounds__(1024) void k_mps2_out(const float* __restrict__ u,
                                                   const float* __restrict__ pack,
                                                   const float* __restrict__ v,
                                                   const float* __restrict__ P,
                                                   float* __restrict__ attn) {
  __shared__ float dots[NH][CS];
  int t = blockIdx.x;
  int c = t >> 4, s0 = c << 4, nIn = t - s0 + 1;
  int tid = threadIdx.x;
  if (tid < 256) {
    int hh = tid >> 4, si = tid & 15;
    int s = s0 + si;
    float d = 0.0f;
    if (s <= t) {
      const float* ur = u    + (size_t)t * (NH * PLEN) + hh * PLEN;
      const float* pr = pack + (size_t)s * (NH * PLEN) + hh * PLEN;
      #pragma unroll
      for (int pp = 0; pp < PLEN; ++pp) d += ur[pp] * pr[pp];
    }
    dots[hh][si] = d;
  }
  __syncthreads();
  int hh = tid >> 6, dd = tid & 63;
  const float* ur = u + (size_t)t * (NH * PLEN) + hh * PLEN;
  const float* Pp = P + (size_t)c * STATE + hh * (PLEN * DH) + dd;
  float a = 0.0f;
  #pragma unroll
  for (int pp = 0; pp < PLEN; ++pp) a += ur[pp] * Pp[pp * DH];
  for (int si = 0; si < nIn; ++si)
    a += dots[hh][si] * v[(size_t)(s0 + si) * DM + hh * DH + dd];
  attn[(size_t)t * DM + hh * DH + dd] = a / (float)(t + 1);
}

// ---------------------------------------------------------------- out = base + LN(y)
__global__ __launch_bounds__(256) void k_add_ln(const float* __restrict__ base,
                                                const float* __restrict__ y,
                                                const float* __restrict__ g,
                                                const float* __restrict__ b,
                                                float* __restrict__ out) {
  __shared__ float red[256];
  int t = blockIdx.x, tid = threadIdx.x;
  const float* yr = y + (size_t)t * DM;
  float vals[4], s = 0.0f, s2 = 0.0f;
  #pragma unroll
  for (int i = 0; i < 4; ++i) {
    float f = yr[tid + 256 * i];
    vals[i] = f; s += f; s2 += f * f;
  }
  red[tid] = s; __syncthreads();
  for (int o = 128; o > 0; o >>= 1) { if (tid < o) red[tid] += red[tid + o]; __syncthreads(); }
  float mu = red[0] / (float)DM;
  __syncthreads();
  red[tid] = s2; __syncthreads();
  for (int o = 128; o > 0; o >>= 1) { if (tid < o) red[tid] += red[tid + o]; __syncthreads(); }
  float var = red[0] / (float)DM - mu * mu;
  float inv = rsqrtf(var + 1e-6f);
  #pragma unroll
  for (int i = 0; i < 4; ++i) {
    int col = tid + 256 * i;
    out[(size_t)t * DM + col] = base[(size_t)t * DM + col] + (vals[i] - mu) * inv * g[col] + b[col];
  }
}

// ---------------------------------------------------------------- host
extern "C" void kernel_launch(void* const* d_in, const int* in_sizes, int n_in,
                              void* d_out, int out_size, void* d_ws, size_t ws_size,
                              hipStream_t stream) {
  (void)in_sizes; (void)n_in; (void)out_size; (void)ws_size;
  const int*   x    = (const int*)  d_in[0];
  const float* emb  = (const float*)d_in[1];
  const float* pos  = (const float*)d_in[2];
  const float* plu  = (const float*)d_in[3];
  const float* wq   = (const float*)d_in[4];
  const float* wk   = (const float*)d_in[5];
  const float* wv   = (const float*)d_in[6];
  const float* wcm  = (const float*)d_in[7];
  const float* bq   = (const float*)d_in[8];
  const float* bk   = (const float*)d_in[9];
  const float* bv   = (const float*)d_in[10];
  const float* bc   = (const float*)d_in[11];
  const float* w1   = (const float*)d_in[12];
  const float* b1   = (const float*)d_in[13];
  const float* w2   = (const float*)d_in[14];
  const float* b2   = (const float*)d_in[15];
  const float* ln1g = (const float*)d_in[16];
  const float* ln1b = (const float*)d_in[17];
  const float* ln2g = (const float*)d_in[18];
  const float* ln2b = (const float*)d_in[19];

  const size_t SD = (size_t)SEQ * DM;         // 2,097,152
  const size_t SP = (size_t)SEQ * NH * PLEN;  // 524,288
  float* ws   = (float*)d_ws;
  float* hb   = ws;            // hidden state
  float* xe   = hb   + SD;
  float* qb   = xe   + SD;     // q, later xr
  float* kb   = qb   + SD;     // k, later attn@wc
  float* vb   = kb   + SD;
  float* atb  = vb   + SD;     // mps2 out, later ff2
  float* pck  = atb  + SD;
  float* unp  = pck  + SP;
  float* ff1  = unp  + SP;     // SEQ*DFF; also aliases chunk-sum buffer
  float* chk  = ff1;           // NC*STATE = 2,097,152 floats (fits in ff1)

  const dim3 gN(SEQ / TM, DM / TN);   // 16 x 8
  const dim3 gF(SEQ / TM, DFF / TN);  // 16 x 32
  const int EW = (int)(SD / 256);     // 8192 elementwise blocks

  k_embed<<<EW, 256, 0, stream>>>(x, emb, hb);

  for (int m = 0; m < LAYERS; ++m) {
    k_addpos<<<EW, 256, 0, stream>>>(hb, pos + (size_t)m * SD, xe);

    k_gemm<<<gN, 256, 0, stream>>>(xe, wq + (size_t)m * DM * DM, bq + (size_t)m * DM,
                                   qb, SEQ, DM, DM, 0.125f, 0);   // * rsqrt(64)
    k_gemm<<<gN, 256, 0, stream>>>(xe, wk + (size_t)m * DM * DM, bk + (size_t)m * DM,
                                   kb, SEQ, DM, DM, 1.0f, 0);
    k_gemm<<<gN, 256, 0, stream>>>(xe, wv + (size_t)m * DM * DM, bv + (size_t)m * DM,
                                   vb, SEQ, DM, DM, 1.0f, 0);

    k_pack<<<SEQ, 256, 0, stream>>>(plu + (size_t)m * PLEN * DM, qb, pck);

    // mps1: x=q, y=k, z=pack  -> unp (pre-softmax)
    k_chunk_outer<<<EW, 256, 0, stream>>>(kb, pck, chk, DH, PLEN);
    k_chunk_scan<<<STATE / 256, 256, 0, stream>>>(chk, NC);
    k_mps1_out<<<SEQ, 256, 0, stream>>>(qb, kb, pck, chk, unp);
    k_softmax16<<<(SEQ * NH) / 256, 256, 0, stream>>>(unp);

    // mps2: x=unpack, y=pack, z=v -> atb
    k_chunk_outer<<<EW, 256, 0, stream>>>(pck, vb, chk, PLEN, DH);
    k_chunk_scan<<<STATE / 256, 256, 0, stream>>>(chk, NC);
    k_mps2_out<<<SEQ, 1024, 0, stream>>>(unp, pck, vb, chk, atb);

    k_gemm<<<gN, 256, 0, stream>>>(atb, wcm + (size_t)m * DM * DM, bc + (size_t)m * DM,
                                   kb, SEQ, DM, DM, 1.0f, 0);     // attn proj -> kb
    k_add_ln<<<SEQ, 256, 0, stream>>>(xe, kb, ln1g + (size_t)m * DM, ln1b + (size_t)m * DM,
                                      qb);                         // xr -> qb

    k_gemm<<<gF, 256, 0, stream>>>(qb, w1 + (size_t)m * DM * DFF, b1 + (size_t)m * DFF,
                                   ff1, SEQ, DFF, DM, 1.0f, 1);    // relu
    k_gemm<<<gN, 256, 0, stream>>>(ff1, w2 + (size_t)m * DFF * DM, b2 + (size_t)m * DM,
                                   atb, SEQ, DM, DFF, 1.0f, 0);    // ff2 -> atb

    float* dst = (m == LAYERS - 1) ? (float*)d_out : hb;
    k_add_ln<<<SEQ, 256, 0, stream>>>(qb, atb, ln2g + (size_t)m * DM, ln2b + (size_t)m * DM, dst);
  }
}
